// RGCN_64149631533101
// MI455X (gfx1250) — compile-verified, matrix operand order checked by reading, not run on hardware
//
#include <hip/hip_runtime.h>

// ---------------------------------------------------------------------------
// Types for CDNA5 WMMA / TDM
// ---------------------------------------------------------------------------
typedef __attribute__((ext_vector_type(16))) __bf16 v16bf;
typedef __attribute__((ext_vector_type(8)))  float  v8f;
typedef __attribute__((ext_vector_type(4)))  unsigned int u32x4;
typedef __attribute__((ext_vector_type(8)))  unsigned int u32x8;

#define NNODE 50000
#define NEDGE 600000

// ---------------------------------------------------------------------------
// helpers
// ---------------------------------------------------------------------------
__device__ __forceinline__ unsigned short f2bf(float f) {
    unsigned int u = __float_as_uint(f);
    u += 0x7FFFu + ((u >> 16) & 1u);   // round-to-nearest-even
    return (unsigned short)(u >> 16);
}

__device__ __forceinline__ unsigned int fkey(float f) {
    unsigned int u = __float_as_uint(f);
    return (u & 0x80000000u) ? ~u : (u | 0x80000000u);
}
__device__ __forceinline__ float fdecode(unsigned int k) {
    unsigned int u = (k & 0x80000000u) ? (k & 0x7FFFFFFFu) : ~k;
    return __uint_as_float(u);
}

// Issue one TDM 2D tile load: tile_rows x 32 bf16 elements, row-padded in LDS
// by 4 dwords every 16 dwords (i.e. LDS row stride = 80 bytes = 20 dwords).
// rows_avail clamps OOB rows to zero-fill; dim0_avail clamps along K.
__device__ __forceinline__ void tdm_load_2d(unsigned lds_byte, const void* gaddr,
                                            int rows_avail, int tile_rows,
                                            int dim0_avail, int K) {
    unsigned long long ga = (unsigned long long)gaddr;
    u32x4 g0;
    g0[0] = 1u;                                            // count=1, user desc
    g0[1] = lds_byte;                                      // lds_addr (bytes)
    g0[2] = (unsigned)(ga & 0xFFFFFFFFull);                // global_addr lo
    g0[3] = (unsigned)((ga >> 32) & 0x1FFFFFFull) | (2u << 30); // addr hi | type=2
    unsigned rows = (unsigned)rows_avail;
    unsigned d0   = (unsigned)dim0_avail;
    u32x8 g1;
    g1[0] = (1u << 16)        // data_size = 2 bytes
          | (1u << 20)        // pad_enable
          | (3u << 22)        // pad_interval: every 64B (16 dwords)
          | (3u << 25);       // pad_amount: 4 dwords (8 bf16)
    g1[1] = (d0 & 0xFFFFu) << 16;                  // tensor_dim0 [63:48]
    g1[2] = (d0 >> 16) | ((rows & 0xFFFFu) << 16); // dim0 hi | tensor_dim1 lo
    g1[3] = (rows >> 16) | (32u << 16);            // dim1 hi | tile_dim0 = 32
    g1[4] = (unsigned)tile_rows;                   // tile_dim1
    g1[5] = (unsigned)K;                           // tensor_dim0_stride lo32
    g1[6] = 0u;
    g1[7] = 0u;
    asm volatile("tensor_load_to_lds %0, %1" :: "s"(g0), "s"(g1) : "memory");
}

// ---------------------------------------------------------------------------
// f32 -> bf16 bulk convert
// ---------------------------------------------------------------------------
__global__ void k_f2bf(const float* __restrict__ x, unsigned short* __restrict__ y,
                       long long n) {
    long long t = (long long)blockIdx.x * blockDim.x + threadIdx.x;
    if (t < n) y[t] = f2bf(x[t]);
}

// zero fill (fp32 / u32 share the bit pattern)
__global__ void k_zero(float* __restrict__ p, long long n) {
    long long t = (long long)blockIdx.x * blockDim.x + threadIdx.x;
    if (t < n) p[t] = 0.0f;
}

// W[K,256] f32  ->  Wt[256,K] bf16
__global__ void k_wtrans(const float* __restrict__ W, unsigned short* __restrict__ Wt,
                         int K) {
    int idx = blockIdx.x * blockDim.x + threadIdx.x;
    if (idx >= K * 256) return;
    int k = idx >> 8, n = idx & 255;
    Wt[(size_t)n * K + k] = f2bf(W[idx]);
}

// ---------------------------------------------------------------------------
// WMMA GEMM with double-buffered TDM tile staging:
//   C[M,256] = A[M,K](bf16) x Wt[256,K](bf16)^T
// Block tile 64x128, 4 waves (2x2); each wave does a 32x64 tile via 2x4
// v_wmma_f32_16x16x32_bf16 per 32-wide K step.  A (64x32) and B (128x32)
// tiles are DMA'd into LDS by the Tensor Data Mover with hardware row
// padding (LDS row stride = 20 dwords).  Two LDS buffer sets: wave 0 issues
// the next tile's DMA, then s_wait_tensorcnt 2 releases the current tile
// (TENSORcnt completes in order), so the next DMA overlaps the WMMA burst.
// ---------------------------------------------------------------------------
__global__ __launch_bounds__(128)
void k_gemm_bf16_wmma(const unsigned short* __restrict__ A,
                      const unsigned short* __restrict__ Bt,
                      float* __restrict__ C,
                      int M, int K) {
    __shared__ unsigned int As[2][64 * 20];    // 64 rows  x (16 data + 4 pad) dw
    __shared__ unsigned int Bs[2][128 * 20];   // 128 rows x (16 data + 4 pad) dw

    const int tid  = threadIdx.x;
    const int wave = tid >> 5;
    const int lane = tid & 31;
    const int wm   = wave >> 1;        // 0..1 : 32-row strip
    const int wn   = wave & 1;         // 0..1 : 64-col strip
    const int half = lane >> 4;        // fragment K-half selector
    const int lrow = lane & 15;        // row (A) / col (B) within 16-tile
    const int bm   = blockIdx.x * 64;
    const int bn   = blockIdx.y * 128;

    const unsigned ldsA[2] = {(unsigned)(unsigned long long)(void*)As[0],
                              (unsigned)(unsigned long long)(void*)As[1]};
    const unsigned ldsB[2] = {(unsigned)(unsigned long long)(void*)Bs[0],
                              (unsigned)(unsigned long long)(void*)Bs[1]};
    const int rowsA = M - bm;          // >= 1, < 65536

    v8f acc[2][4] = {};

    union Frag { uint4 q[2]; v16bf v; };

    const int nsteps = K >> 5;

    // prologue: start DMA of tile 0 into buffer 0
    if (tid < 32) {
        tdm_load_2d(ldsA[0], A + (size_t)bm * K, rowsA, 64,  K, K);
        tdm_load_2d(ldsB[0], Bt + (size_t)bn * K, 128,   128, K, K);
    }

    for (int step = 0; step < nsteps; ++step) {
        const int cur = step & 1;
        const int nxt = cur ^ 1;
        if (tid < 32) {
            int k1 = (step + 1) << 5;
            if (k1 < K) {
                // buffer `nxt` was last read in step-1; the end-of-iteration
                // barrier below guarantees all waves are done with it.
                tdm_load_2d(ldsA[nxt], A + (size_t)bm * K + k1, rowsA, 64,
                            K - k1, K);
                tdm_load_2d(ldsB[nxt], Bt + (size_t)bn * K + k1, 128, 128,
                            K - k1, K);
                __builtin_amdgcn_s_wait_tensorcnt(2);  // current tile complete
            } else {
                __builtin_amdgcn_s_wait_tensorcnt(0);  // last tile complete
            }
        }
        __syncthreads();               // current tile visible to all waves

        const unsigned int* Ab = As[cur];
        const unsigned int* Bb = Bs[cur];

        // fragment loads (ds_load_b128 pairs in the ISA 16-bit A/B layout)
        Frag fa[2], fb[4];
#pragma unroll
        for (int t = 0; t < 2; ++t) {
            int ar = wm * 32 + t * 16 + lrow;
            fa[t].q[0] = *(const uint4*)(&Ab[ar * 20 + half * 4]);
            fa[t].q[1] = *(const uint4*)(&Ab[ar * 20 + 8 + half * 4]);
        }
#pragma unroll
        for (int j = 0; j < 4; ++j) {
            int br = wn * 64 + j * 16 + lrow;
            fb[j].q[0] = *(const uint4*)(&Bb[br * 20 + half * 4]);
            fb[j].q[1] = *(const uint4*)(&Bb[br * 20 + 8 + half * 4]);
        }
#pragma unroll
        for (int i = 0; i < 2; ++i)
#pragma unroll
            for (int j = 0; j < 4; ++j)
                acc[i][j] = __builtin_amdgcn_wmma_f32_16x16x32_bf16(
                    false, fa[i].v, false, fb[j].v,
                    (short)0, acc[i][j], false, false);

        __syncthreads();               // all waves done reading `cur` before
                                       // it is overwritten by the next step's DMA
    }

    // epilogue: C/D layout -> lane n = lrow, row m = v + half*8
#pragma unroll
    for (int i = 0; i < 2; ++i) {
        int mBase = bm + wm * 32 + i * 16 + half * 8;
#pragma unroll
        for (int j = 0; j < 4; ++j) {
            int n = bn + wn * 64 + j * 16 + lrow;
#pragma unroll
            for (int v = 0; v < 8; ++v) {
                int m = mBase + v;
                if (m < M) C[(size_t)m * 256 + n] = acc[i][j][v];
            }
        }
    }
}

// ---------------------------------------------------------------------------
// per-(node,head) attention projections:  out[m,h] = sum_d z[m, h*64+d]*av[h,d]
// ---------------------------------------------------------------------------
__global__ void k_head_dot(const float* __restrict__ z, const float* __restrict__ av,
                           float* __restrict__ out, int M) {
    int t = blockIdx.x * blockDim.x + threadIdx.x;
    if (t >= M * 4) return;
    int m = t >> 2, h = t & 3;
    const float* zp = z + (size_t)m * 256 + h * 64;
    const float* ap = av + h * 64;
    float s = 0.f;
#pragma unroll 4
    for (int d = 0; d < 64; d += 4) {
        float4 zz = *(const float4*)(zp + d);
        float4 aa = *(const float4*)(ap + d);
        s += zz.x * aa.x + zz.y * aa.y + zz.z * aa.z + zz.w * aa.w;
    }
    out[t] = s;
}

// edge logits + segment max (monotone-uint atomicMax)
__global__ void k_edge_logits(const int* __restrict__ src, const int* __restrict__ dst,
                              const float* __restrict__ el, const float* __restrict__ er,
                              float* __restrict__ ebuf, unsigned int* __restrict__ mkey,
                              int E) {
    int t = blockIdx.x * blockDim.x + threadIdx.x;
    if (t >= E * 4) return;
    int e = t >> 2, h = t & 3;
    float v = el[src[e] * 4 + h] + er[dst[e] * 4 + h];
    v = v > 0.f ? v : 0.2f * v;          // leaky_relu(0.2)
    ebuf[t] = v;
    atomicMax(&mkey[dst[e] * 4 + h], fkey(v));
}

// exp(e - max) + segment sum
__global__ void k_edge_exp(const int* __restrict__ dst, float* __restrict__ ebuf,
                           const unsigned int* __restrict__ mkey,
                           float* __restrict__ s, int E) {
    int t = blockIdx.x * blockDim.x + threadIdx.x;
    if (t >= E * 4) return;
    int e = t >> 2, h = t & 3;
    float m  = fdecode(mkey[dst[e] * 4 + h]);
    float ex = __expf(ebuf[t] - m);
    ebuf[t]  = ex;
    atomicAdd(&s[dst[e] * 4 + h], ex);
}

// out[dst,:] += z[src,:] * alpha  (thread = edge x 4-channel group)
__global__ void k_edge_agg(const int* __restrict__ src, const int* __restrict__ dst,
                           const float* __restrict__ ebuf, const float* __restrict__ s,
                           const float* __restrict__ z, float* __restrict__ out, int E) {
    long long t = (long long)blockIdx.x * blockDim.x + threadIdx.x;
    if (t >= (long long)E * 64) return;
    int e  = (int)(t >> 6);
    int cg = ((int)t & 63) << 2;          // channel group base, 0..252
    int h  = cg >> 6;                     // head
    int d  = dst[e], si = src[e];
    float alpha = ebuf[e * 4 + h] / s[d * 4 + h];
    float4 zz = *(const float4*)(z + (size_t)si * 256 + cg);
    float* op = out + (size_t)d * 256 + cg;
    atomicAdd(op + 0, zz.x * alpha);
    atomicAdd(op + 1, zz.y * alpha);
    atomicAdd(op + 2, zz.z * alpha);
    atomicAdd(op + 3, zz.w * alpha);
}

// ---------------------------------------------------------------------------
// batchnorm: per-channel sum / sumsq  (thread = channel, 128 rows per block)
// ---------------------------------------------------------------------------
__global__ void k_bn_stats(const float* __restrict__ x, float* __restrict__ sum,
                           float* __restrict__ sumsq, int M) {
    int c  = threadIdx.x;                 // 0..255
    int r0 = blockIdx.x * 128;
    float s = 0.f, q = 0.f;
    for (int r = 0; r < 128; ++r) {
        int m = r0 + r;
        if (m < M) {
            float v = x[(size_t)m * 256 + c];
            s += v; q += v * v;
        }
    }
    atomicAdd(&sum[c], s);
    atomicAdd(&sumsq[c], q);
}

// bn apply + activation, emit bf16 for next layer's GEMM
__global__ void k_bn_apply(const float* __restrict__ x, const float* __restrict__ sum,
                           const float* __restrict__ sumsq,
                           const float* __restrict__ g, const float* __restrict__ beta,
                           unsigned short* __restrict__ out_bf, int M, int act_tanh) {
    long long t = (long long)blockIdx.x * blockDim.x + threadIdx.x;
    if (t >= (long long)M * 256) return;
    int c = (int)t & 255;
    float invM = 1.0f / (float)M;
    float mu  = sum[c] * invM;
    float var = sumsq[c] * invM - mu * mu;
    float y = (x[t] - mu) * rsqrtf(var + 1e-5f) * g[c] + beta[c];
    y = act_tanh ? tanhf(y) : (y > 0.f ? y : 0.01f * y);
    out_bf[t] = f2bf(y);
}

// final layer: out = x + bias (fp32)
__global__ void k_add_bias_out(const float* __restrict__ x, const float* __restrict__ b,
                               float* __restrict__ out, int M) {
    long long t = (long long)blockIdx.x * blockDim.x + threadIdx.x;
    if (t >= (long long)M * 256) return;
    out[t] = x[t] + b[(int)t & 255];
}

// ---------------------------------------------------------------------------
// host launcher
// ---------------------------------------------------------------------------
extern "C" void kernel_launch(void* const* d_in, const int* in_sizes, int n_in,
                              void* d_out, int out_size, void* d_ws, size_t ws_size,
                              hipStream_t stream) {
    const int M = NNODE, E = NEDGE;
    const float* x_user = (const float*)d_in[0];
    const float* x_item = (const float*)d_in[1];
    const int* ui_src = (const int*)d_in[2];
    const int* ui_dst = (const int*)d_in[3];
    const int* iu_src = (const int*)d_in[4];
    const int* iu_dst = (const int*)d_in[5];
    const float* Wl[4]  = {(const float*)d_in[6],  (const float*)d_in[10],
                           (const float*)d_in[14], (const float*)d_in[18]};
    const float* alL[4] = {(const float*)d_in[7],  (const float*)d_in[11],
                           (const float*)d_in[15], (const float*)d_in[19]};
    const float* arL[4] = {(const float*)d_in[8],  (const float*)d_in[12],
                           (const float*)d_in[16], (const float*)d_in[20]};
    const float* bL[4]  = {(const float*)d_in[9],  (const float*)d_in[13],
                           (const float*)d_in[17], (const float*)d_in[21]};
    const float* bn_g = (const float*)d_in[22];   // [3,2,256]
    const float* bn_b = (const float*)d_in[23];

    char* ws = (char*)d_ws;
    size_t off = 0;
    auto alloc = [&](size_t bytes) -> void* {
        void* p = ws + off;
        off += (bytes + 255) & ~(size_t)255;
        return p;
    };
    unsigned short* hu_bf = (unsigned short*)alloc((size_t)M * 256 * 2);
    unsigned short* hi_bf = (unsigned short*)alloc((size_t)M * 256 * 2);
    float* z0 = (float*)alloc((size_t)M * 256 * 4);
    float* z1 = (float*)alloc((size_t)M * 256 * 4);
    float* z2 = (float*)alloc((size_t)M * 256 * 4);
    float* z3 = (float*)alloc((size_t)M * 256 * 4);
    float* newU = (float*)alloc((size_t)M * 256 * 4);
    float* newI = (float*)alloc((size_t)M * 256 * 4);
    unsigned short* Wt0 = (unsigned short*)alloc((size_t)256 * 256 * 2);
    unsigned short* Wt1 = (unsigned short*)alloc((size_t)256 * 256 * 2);
    float* elb = (float*)alloc((size_t)M * 4 * 4);
    float* erb = (float*)alloc((size_t)M * 4 * 4);
    unsigned int* mkey = (unsigned int*)alloc((size_t)M * 4 * 4);
    float* sbuf = (float*)alloc((size_t)M * 4 * 4);
    float* ebuf = (float*)alloc((size_t)E * 4 * 4);
    float* bnsum = (float*)alloc(256 * 4);
    float* bnsq  = (float*)alloc(256 * 4);

    // initial feature convert (layer-1 K=128)
    {
        long long n = (long long)M * 128;
        int g = (int)((n + 255) / 256);
        k_f2bf<<<g, 256, 0, stream>>>(x_user, hu_bf, n);
        k_f2bf<<<g, 256, 0, stream>>>(x_item, hi_bf, n);
    }

    const int dims[4] = {128, 256, 256, 256};
    const int gN4   = (M * 4 + 255) / 256;
    const int gE4   = (E * 4 + 255) / 256;
    const int gE64  = (int)(((long long)E * 64 + 255) / 256);
    const int gM256 = (int)(((long long)M * 256 + 255) / 256);

    for (int l = 0; l < 4; ++l) {
        const int K = dims[l];
        // weight transpose+convert
        k_wtrans<<<(K * 256 + 255) / 256, 256, 0, stream>>>(Wl[l], Wt0, K);
        k_wtrans<<<(K * 256 + 255) / 256, 256, 0, stream>>>(Wl[l] + (size_t)K * 256, Wt1, K);

        dim3 gg((M + 63) / 64, 2);
        k_gemm_bf16_wmma<<<gg, 128, 0, stream>>>(hu_bf, Wt0, z0, M, K); // z_src (ui)
        k_gemm_bf16_wmma<<<gg, 128, 0, stream>>>(hi_bf, Wt0, z1, M, K); // z_dst (ui)
        k_gemm_bf16_wmma<<<gg, 128, 0, stream>>>(hi_bf, Wt1, z2, M, K); // z_src (iu)
        k_gemm_bf16_wmma<<<gg, 128, 0, stream>>>(hu_bf, Wt1, z3, M, K); // z_dst (iu)

        // ---- relation user->item, output newI ----
        k_head_dot<<<gN4, 256, 0, stream>>>(z0, alL[l] + 0,   elb, M);
        k_head_dot<<<gN4, 256, 0, stream>>>(z1, arL[l] + 0,   erb, M);
        k_zero<<<gN4, 256, 0, stream>>>((float*)mkey, (long long)M * 4);
        k_zero<<<gN4, 256, 0, stream>>>(sbuf, (long long)M * 4);
        k_zero<<<gM256, 256, 0, stream>>>(newI, (long long)M * 256);
        k_edge_logits<<<gE4, 256, 0, stream>>>(ui_src, ui_dst, elb, erb, ebuf, mkey, E);
        k_edge_exp<<<gE4, 256, 0, stream>>>(ui_dst, ebuf, mkey, sbuf, E);
        k_edge_agg<<<gE64, 256, 0, stream>>>(ui_src, ui_dst, ebuf, sbuf, z0, newI, E);

        // ---- relation item->user, output newU ----
        k_head_dot<<<gN4, 256, 0, stream>>>(z2, alL[l] + 256, elb, M);
        k_head_dot<<<gN4, 256, 0, stream>>>(z3, arL[l] + 256, erb, M);
        k_zero<<<gN4, 256, 0, stream>>>((float*)mkey, (long long)M * 4);
        k_zero<<<gN4, 256, 0, stream>>>(sbuf, (long long)M * 4);
        k_zero<<<gM256, 256, 0, stream>>>(newU, (long long)M * 256);
        k_edge_logits<<<gE4, 256, 0, stream>>>(iu_src, iu_dst, elb, erb, ebuf, mkey, E);
        k_edge_exp<<<gE4, 256, 0, stream>>>(iu_dst, ebuf, mkey, sbuf, E);
        k_edge_agg<<<gE64, 256, 0, stream>>>(iu_src, iu_dst, ebuf, sbuf, z2, newU, E);

        if (l < 3) {
            // bias is per-channel constant -> cancels inside batchnorm; skip it.
            int act = (l == 2) ? 1 : 0;   // layer 3 uses tanh, else leaky(0.01)
            // user
            k_zero<<<1, 256, 0, stream>>>(bnsum, 256);
            k_zero<<<1, 256, 0, stream>>>(bnsq, 256);
            k_bn_stats<<<(M + 127) / 128, 256, 0, stream>>>(newU, bnsum, bnsq, M);
            k_bn_apply<<<gM256, 256, 0, stream>>>(newU, bnsum, bnsq,
                                                  bn_g + (size_t)(l * 2 + 0) * 256,
                                                  bn_b + (size_t)(l * 2 + 0) * 256,
                                                  hu_bf, M, act);
            // item
            k_zero<<<1, 256, 0, stream>>>(bnsum, 256);
            k_zero<<<1, 256, 0, stream>>>(bnsq, 256);
            k_bn_stats<<<(M + 127) / 128, 256, 0, stream>>>(newI, bnsum, bnsq, M);
            k_bn_apply<<<gM256, 256, 0, stream>>>(newI, bnsum, bnsq,
                                                  bn_g + (size_t)(l * 2 + 1) * 256,
                                                  bn_b + (size_t)(l * 2 + 1) * 256,
                                                  hi_bf, M, act);
        } else {
            float* out = (float*)d_out;
            k_add_bias_out<<<gM256, 256, 0, stream>>>(newU, bL[3] + 256, out, M);
            k_add_bias_out<<<gM256, 256, 0, stream>>>(newI, bL[3] + 0,
                                                      out + (size_t)M * 256, M);
        }
    }
}
